// SurrogateGIN_85985245266464
// MI455X (gfx1250) — matrix operand order repaired and checked by reading, not compiled, not run on hardware
//
#include <hip/hip_runtime.h>

#define NN 50000
#define NE 800000
#define INC 128
#define HID 256
#define OUTC 40

typedef __attribute__((ext_vector_type(16))) __bf16 v16bf;
typedef __attribute__((ext_vector_type(8)))  float  v8f;

// ---------- helpers ----------
static __device__ __forceinline__ unsigned short f2bf(float f) {
  unsigned int u = __float_as_uint(f);
  u = (u + 0x7FFFu + ((u >> 16) & 1u)) >> 16;
  return (unsigned short)u;
}
static __device__ __forceinline__ unsigned int pack2bf(float a, float b) {
  return (unsigned int)f2bf(a) | ((unsigned int)f2bf(b) << 16);
}

// ---------- simple data-movement kernels ----------
__global__ void gin_copy_f32(const float* __restrict__ in, float* __restrict__ out, int n4) {
  int i = blockIdx.x * blockDim.x + threadIdx.x;
  if (i < n4) ((float4*)out)[i] = ((const float4*)in)[i];
}

__global__ void gin_conv_bf16(const float* __restrict__ in, unsigned short* __restrict__ out, int n2) {
  int i = blockIdx.x * blockDim.x + threadIdx.x;
  if (i < n2) ((unsigned int*)out)[i] = pack2bf(in[2 * i], in[2 * i + 1]);
}

// Wlin [512][40] f32 -> [512][64] bf16, zero padded
__global__ void gin_pad_wlin_bf16(const float* __restrict__ in, unsigned short* __restrict__ out) {
  int i = blockIdx.x * blockDim.x + threadIdx.x;
  if (i >= 512 * 64) return;
  int r = i >> 6, c = i & 63;
  float v = (c < OUTC) ? in[r * OUTC + c] : 0.0f;
  out[i] = f2bf(v);
}

// z[dst] += h[src] over edges; z pre-initialized with h (eps=0 GIN)
__global__ void gin_scatter_add(const float* __restrict__ h,
                                const long long* __restrict__ ei,
                                float* __restrict__ z, int C4, int C) {
  int i = blockIdx.x * blockDim.x + threadIdx.x;
  if (i >= NE * C4) return;
  int e = i / C4, c4 = i - e * C4;
  int src = (int)ei[e];
  int dst = (int)ei[NE + e];
  float4 v = ((const float4*)(h + (long long)src * C))[c4];
  float* zp = z + (long long)dst * C + c4 * 4;
  atomicAdd(zp + 0, v.x);
  atomicAdd(zp + 1, v.y);
  atomicAdd(zp + 2, v.z);
  atomicAdd(zp + 3, v.w);
}

// ---------- WMMA GEMM: C[M,ldC] = act( [A|A2][M,K] * Bbf[K,ldB] + bias ) ----------
// Double-buffered LDS; B tiles staged via GLOBAL_LOAD_ASYNC_TO_LDS_B128 (ASYNCcnt),
// A tiles prefetched to registers (f32->bf16 conversion) and stored after compute.
#define TM 128
#define TN 64
#define TK 32
#define ASTR 40   // bf16 elems; 80B rows, keeps 16B alignment for b128 frag loads
#define BSTR 80   // bf16 elems; 160B rows

__global__ __launch_bounds__(256)
void gin_gemm_bf16(const float* __restrict__ A, const float* __restrict__ A2,
                   const unsigned short* __restrict__ B,
                   const float* __restrict__ bias, float* __restrict__ C,
                   int M, int K, int kSplit, int ldA, int ldB, int ldC,
                   int biasN, int doRelu) {
  __shared__ __align__(16) unsigned short Alds[2][TM * ASTR];
  __shared__ __align__(16) unsigned short Blds[2][TK * BSTR];

  const int tid  = threadIdx.x;
  const int lane = tid & 31;
  const int wave = tid >> 5;                 // 0..7, each wave: 16 rows x 64 cols
  const int rowBlock = blockIdx.x * TM;
  const int colBlock = blockIdx.y * TN;

  const int br = tid >> 3;                   // B stage: row 0..31
  const int bg = tid & 7;                    // B stage: 8 bf16 per thread

  v8f acc[4] = {};
  float4 aregs[4];

  auto loadA = [&](int k0) {
#pragma unroll
    for (int p = 0; p < 4; ++p) {
      int idx = tid + p * 256;               // 0..1023
      int r = idx >> 3, c4 = idx & 7;
      int grow = rowBlock + r;
      float4 v = make_float4(0.f, 0.f, 0.f, 0.f);
      if (grow < M) {
        int kk = k0 + c4 * 4;
        const float* ap = (kk < kSplit) ? (A  + (long long)grow * ldA + kk)
                                        : (A2 + (long long)grow * ldA + (kk - kSplit));
        v = *(const float4*)ap;
      }
      aregs[p] = v;
    }
  };
  auto storeA = [&](int buf) {
#pragma unroll
    for (int p = 0; p < 4; ++p) {
      int idx = tid + p * 256;
      int r = idx >> 3, c4 = idx & 7;
      unsigned int* dp = (unsigned int*)&Alds[buf][r * ASTR + c4 * 4];
      dp[0] = pack2bf(aregs[p].x, aregs[p].y);
      dp[1] = pack2bf(aregs[p].z, aregs[p].w);
    }
  };
  auto issueB = [&](int k0, int buf) {
    // DMA 16B/lane of bf16 weights straight into LDS (no VGPR round-trip).
    unsigned loff = (unsigned)(size_t)&Blds[buf][br * BSTR + bg * 8];
    unsigned goff = (unsigned)(((k0 + br) * ldB + colBlock + bg * 8) * 2);
    asm volatile("global_load_async_to_lds_b128 %0, %1, %2"
                 :: "v"(loff), "v"(goff), "s"(B) : "memory");
  };

  // prologue: chunk 0 in flight / staged into buffer 0
  issueB(0, 0);
  loadA(0);
  storeA(0);

  const int nCh = K / TK;
  for (int i = 0; i < nCh; ++i) {
    const int cur = i & 1;
    const bool more = (i + 1 < nCh);
    if (more) {
      issueB((i + 1) * TK, cur ^ 1);         // DMA next B tile into other buffer
      loadA((i + 1) * TK);                   // prefetch next A tile to registers
    }
    // async loads complete in order; <=1 outstanding means chunk i has landed
    if (more) asm volatile("s_wait_asynccnt 0x1" ::: "memory");
    else      asm volatile("s_wait_asynccnt 0x0" ::: "memory");
    __syncthreads();

    // A fragment (ISA 16-bit A layout): lanes<16 -> K{0..7,16..23}, lanes>=16 -> K{8..15,24..31}
    union Frag { uint4 u[2]; v16bf v; };
    Frag af;
    {
      int r = wave * 16 + (lane & 15);
      int kOff = (lane < 16) ? 0 : 8;
      const unsigned short* abase = &Alds[cur][r * ASTR + kOff];
      af.u[0] = *(const uint4*)abase;
      af.u[1] = *(const uint4*)(abase + 16);
    }
    // Load all 4 B fragments into distinct registers first so the ds_load ->
    // wmma chain is not serialized by a full dscnt drain per WMMA.
    Frag bfr[4];
#pragma unroll
    for (int t = 0; t < 4; ++t) {
      const int bkOff = (lane < 16) ? 0 : 8;
      const unsigned short* bp = &Blds[cur][bkOff * BSTR + t * 16 + (lane & 15)];
      unsigned short* bs = (unsigned short*)&bfr[t];
#pragma unroll
      for (int s = 0; s < 8; ++s) {
        bs[s]     = bp[s * BSTR];
        bs[s + 8] = bp[(16 + s) * BSTR];
      }
    }
#pragma unroll
    for (int t = 0; t < 4; ++t) {
      acc[t] = __builtin_amdgcn_wmma_f32_16x16x32_bf16(
          false, af.v, false, bfr[t].v, (short)0, acc[t], false, false);
    }
    if (more) storeA(cur ^ 1);               // write prefetched A into other buffer
    __syncthreads();
  }

  // epilogue: C/D layout -> VGPR r is M=r (lanes<16) or M=r+8 (lanes>=16), lane%16 is N
  int rBase = rowBlock + wave * 16 + ((lane < 16) ? 0 : 8);
  int col = lane & 15;
#pragma unroll
  for (int t = 0; t < 4; ++t) {
    int gcol = colBlock + t * 16 + col;
    float bv = (gcol < biasN) ? bias[gcol] : 0.0f;
#pragma unroll
    for (int r = 0; r < 8; ++r) {
      int grow = rBase + r;
      if (grow < M) {
        float v = acc[t][r] + bv;
        if (doRelu) v = fmaxf(v, 0.0f);
        C[(long long)grow * ldC + gcol] = v;
      }
    }
  }
}

// ---------- log-softmax over 40 classes, one wave per node ----------
__global__ void gin_log_softmax(const float* __restrict__ logits, float* __restrict__ out, int n) {
  int lane = threadIdx.x & 31;
  int wave = threadIdx.x >> 5;
  int node = blockIdx.x * (blockDim.x >> 5) + wave;
  if (node >= n) return;
  const float* row = logits + (long long)node * 64;
  float ninf = -__builtin_inff();
  float v0 = (lane < OUTC) ? row[lane] : ninf;
  float v1 = (lane + 32 < OUTC) ? row[lane + 32] : ninf;
  float m = fmaxf(v0, v1);
#pragma unroll
  for (int off = 16; off; off >>= 1) m = fmaxf(m, __shfl_xor(m, off, 32));
  float s = ((lane < OUTC) ? __expf(v0 - m) : 0.0f) +
            ((lane + 32 < OUTC) ? __expf(v1 - m) : 0.0f);
#pragma unroll
  for (int off = 16; off; off >>= 1) s += __shfl_xor(s, off, 32);
  float ls = __logf(s);
  if (lane < OUTC) out[(long long)node * OUTC + lane] = v0 - m - ls;
  if (lane + 32 < OUTC) out[(long long)node * OUTC + lane + 32] = v1 - m - ls;
}

// ---------- host launch ----------
extern "C" void kernel_launch(void* const* d_in, const int* in_sizes, int n_in,
                              void* d_out, int out_size, void* d_ws, size_t ws_size,
                              hipStream_t stream) {
  const float*     x    = (const float*)d_in[0];
  const long long* ei   = (const long long*)d_in[1];
  const float*     W1a  = (const float*)d_in[2];
  const float*     b1a  = (const float*)d_in[3];
  const float*     W2a  = (const float*)d_in[4];
  const float*     b2a  = (const float*)d_in[5];
  const float*     W1b  = (const float*)d_in[6];
  const float*     b1b  = (const float*)d_in[7];
  const float*     W2b  = (const float*)d_in[8];
  const float*     b2b  = (const float*)d_in[9];
  const float*     Wlin = (const float*)d_in[10];
  const float*     blin = (const float*)d_in[11];

  char* ws = (char*)d_ws;
  size_t off = 0;
  auto carve = [&](size_t bytes) -> char* {
    char* p = ws + off;
    off = (off + bytes + 255) & ~(size_t)255;
    return p;
  };
  const size_t nodeBuf = (size_t)NN * HID * sizeof(float);
  float* zbuf   = (float*)carve(nodeBuf);
  float* tbuf   = (float*)carve(nodeBuf);
  float* h1     = (float*)carve(nodeBuf);
  float* h2     = (float*)carve(nodeBuf);
  float* logits = (float*)carve((size_t)NN * 64 * sizeof(float));
  unsigned short* w1a_bf  = (unsigned short*)carve(INC * HID * 2);
  unsigned short* w2a_bf  = (unsigned short*)carve(HID * HID * 2);
  unsigned short* w1b_bf  = (unsigned short*)carve(HID * HID * 2);
  unsigned short* w2b_bf  = (unsigned short*)carve(HID * HID * 2);
  unsigned short* wlin_bf = (unsigned short*)carve(512 * 64 * 2);

  dim3 blk(256);
  auto grid1 = [](int n) { return dim3((unsigned)((n + 255) / 256)); };

  // weight conversions (bf16)
  gin_conv_bf16<<<grid1(INC * HID / 2), blk, 0, stream>>>(W1a, w1a_bf, INC * HID / 2);
  gin_conv_bf16<<<grid1(HID * HID / 2), blk, 0, stream>>>(W2a, w2a_bf, HID * HID / 2);
  gin_conv_bf16<<<grid1(HID * HID / 2), blk, 0, stream>>>(W1b, w1b_bf, HID * HID / 2);
  gin_conv_bf16<<<grid1(HID * HID / 2), blk, 0, stream>>>(W2b, w2b_bf, HID * HID / 2);
  gin_pad_wlin_bf16<<<grid1(512 * 64), blk, 0, stream>>>(Wlin, wlin_bf);

  const dim3 gemmGridFull((NN + TM - 1) / TM, HID / TN);   // 391 x 4
  const dim3 gemmGridLin((NN + TM - 1) / TM, 1);           // 391 x 1

  // ----- layer a -----
  gin_copy_f32<<<grid1(NN * INC / 4), blk, 0, stream>>>(x, zbuf, NN * INC / 4);
  gin_scatter_add<<<grid1(NE * (INC / 4)), blk, 0, stream>>>(x, ei, zbuf, INC / 4, INC);
  gin_gemm_bf16<<<gemmGridFull, blk, 0, stream>>>(zbuf, zbuf, w1a_bf, b1a, tbuf,
                                                  NN, INC, INC, INC, HID, HID, HID, 1);
  gin_gemm_bf16<<<gemmGridFull, blk, 0, stream>>>(tbuf, tbuf, w2a_bf, b2a, h1,
                                                  NN, HID, HID, HID, HID, HID, HID, 1);
  // ----- layer b -----
  gin_copy_f32<<<grid1(NN * HID / 4), blk, 0, stream>>>(h1, zbuf, NN * HID / 4);
  gin_scatter_add<<<grid1(NE * (HID / 4)), blk, 0, stream>>>(h1, ei, zbuf, HID / 4, HID);
  gin_gemm_bf16<<<gemmGridFull, blk, 0, stream>>>(zbuf, zbuf, w1b_bf, b1b, tbuf,
                                                  NN, HID, HID, HID, HID, HID, HID, 1);
  gin_gemm_bf16<<<gemmGridFull, blk, 0, stream>>>(tbuf, tbuf, w2b_bf, b2b, h2,
                                                  NN, HID, HID, HID, HID, HID, HID, 1);
  // ----- readout: [h1 | h2] @ Wlin + blin -----
  gin_gemm_bf16<<<gemmGridLin, blk, 0, stream>>>(h1, h2, wlin_bf, blin, logits,
                                                 NN, 2 * HID, HID, HID, 64, 64, OUTC, 0);
  // ----- log-softmax -----
  gin_log_softmax<<<dim3((NN + 7) / 8), blk, 0, stream>>>(logits, (float*)d_out, NN);
}
